// SelfAttention_18287970746865
// MI455X (gfx1250) — compile-verified, matrix operand order checked by reading, not run on hardware
//
#include <hip/hip_runtime.h>

typedef __bf16 bf16;
typedef __attribute__((ext_vector_type(16))) __bf16 v16bf;
typedef __attribute__((ext_vector_type(8)))  __bf16 v8bf;
typedef __attribute__((ext_vector_type(4)))  __bf16 v4bf;
typedef __attribute__((ext_vector_type(8)))  float  v8f;
typedef __attribute__((vector_size(16)))     int    v4i_;

#define WMMA_BF16(A, B, C) \
  __builtin_amdgcn_wmma_f32_16x16x32_bf16(false, (A), false, (B), (short)0, (C), false, false)

#if __has_builtin(__builtin_amdgcn_global_load_async_to_lds_b128)
#define HAVE_ASYNC 1
#else
#define HAVE_ASYNC 0
#endif

// 16-byte global -> LDS stage: async (no VGPR round-trip, ASYNCcnt) if available.
__device__ __forceinline__ void stage16(bf16* dst_lds, const bf16* src_glb) {
#if HAVE_ASYNC
  __builtin_amdgcn_global_load_async_to_lds_b128(
      (__attribute__((address_space(1))) v4i_*)(src_glb),
      (__attribute__((address_space(3))) v4i_*)(dst_lds),
      0, 0);
#else
  *(uint4*)dst_lds = *(const uint4*)src_glb;
#endif
}

__device__ __forceinline__ void wait_async0() {
#if HAVE_ASYNC
#if __has_builtin(__builtin_amdgcn_s_wait_asynccnt)
  __builtin_amdgcn_s_wait_asynccnt(0);
#else
  asm volatile("s_wait_asynccnt 0x0" ::: "memory");
#endif
#endif
}

__device__ __forceinline__ v16bf ld_frag(const bf16* p0, const bf16* p1) {
  v8bf lo = *(const v8bf*)p0;
  v8bf hi = *(const v8bf*)p1;
  return __builtin_shufflevector(lo, hi, 0,1,2,3,4,5,6,7,8,9,10,11,12,13,14,15);
}

// ---------------------------------------------------------------- convert X
__global__ __launch_bounds__(256)
void cvt_x_kernel(const float* __restrict__ X, bf16* __restrict__ Xbf, int n) {
  int i = (blockIdx.x * 256 + threadIdx.x) * 4;
  if (i < n) {
    float4 f = *(const float4*)(X + i);
    v4bf o;
    o[0] = (bf16)f.x; o[1] = (bf16)f.y; o[2] = (bf16)f.z; o[3] = (bf16)f.w;
    *(v4bf*)(Xbf + i) = o;
  }
}

// ------------------------------------------- convert + transpose Wq/Wk/Wv
__global__ __launch_bounds__(256)
void cvt_w_kernel(const float* __restrict__ Wq, const float* __restrict__ Wk,
                  const float* __restrict__ Wv, bf16* __restrict__ Wt) {
  constexpr int E = 1024;
  const int z = blockIdx.z;
  const float* W = (z == 0) ? Wq : (z == 1) ? Wk : Wv;
  bf16* out = Wt + (size_t)z * E * E;
  __shared__ float tile[32][33];
  const int k0 = blockIdx.y * 32;
  const int n0 = blockIdx.x * 32;
  const int t  = threadIdx.x;
  const int r  = t >> 3;
  const int c4 = (t & 7) * 4;
#pragma unroll
  for (int e = 0; e < 4; ++e)
    tile[r][c4 + e] = W[(size_t)(k0 + r) * E + n0 + c4 + e];
  __syncthreads();
#pragma unroll
  for (int e = 0; e < 4; ++e)  // out[n][k] = W[k][n]
    out[(size_t)(n0 + r) * E + k0 + c4 + e] = (bf16)tile[c4 + e][r];
}

// --------------------------------------------------- QKV projection GEMM
// C[8192,1024] = Xbf @ W + bias ; z=0->Q (pre-scaled 1/8, [B,H,S,D]),
// z=1->K ([B,H,S,D]), z=2->V stored transposed ([B,H,D,S]).
// Block tile 64x256, 8 waves, wave tile 32x64 (8 WMMA / k-step / wave).
__global__ __launch_bounds__(256)
void qkv_gemm_kernel(const bf16* __restrict__ Xbf, const bf16* __restrict__ Wt,
                     const float* __restrict__ bq, const float* __restrict__ bk,
                     const float* __restrict__ bvv,
                     bf16* __restrict__ Qw, bf16* __restrict__ Kw, bf16* __restrict__ Vtw) {
  constexpr int E = 1024;
  constexpr int NSTEP = E / 32;
  const int z = blockIdx.z;
  const bf16* W = Wt + (size_t)z * E * E;               // W^T: [N][K]
  const float* bias = (z == 0) ? bq : (z == 1) ? bk : bvv;

  const int n0 = blockIdx.x * 256;
  const int m0 = blockIdx.y * 64;

  __shared__ __align__(16) bf16 As[2][64 * 32];    //  8 KB
  __shared__ __align__(16) bf16 Bs[2][256 * 32];   // 32 KB

  const int tid  = threadIdx.x;
  const int wave = tid >> 5;
  const int lane = tid & 31;
  const int wm = wave & 1;        // 2 M-groups of 32 rows
  const int wn = wave >> 1;       // 4 N-groups of 64 cols
  const int ra = lane & 15;
  const int kg = (lane >> 4) * 8;
  const int half = lane >> 4;

  auto stage = [&](int buf, int k0) {
    {  // A: 64x32 bf16 = 256 x 16B chunks, 1/thread
      int row = tid >> 2, ch = tid & 3;
      stage16(&As[buf][row * 32 + ch * 8],
              &Xbf[(size_t)(m0 + row) * E + k0 + ch * 8]);
    }
#pragma unroll
    for (int i = 0; i < 4; ++i) {  // B: 256x32 bf16 = 1024 chunks, 4/thread
      int c2 = tid + 256 * i;
      int r2 = c2 >> 2, ch2 = c2 & 3;
      stage16(&Bs[buf][r2 * 32 + ch2 * 8],
              &W[(size_t)(n0 + r2) * E + k0 + ch2 * 8]);
    }
  };

  v8f zero = {};
  v8f acc[2][4];
#pragma unroll
  for (int i = 0; i < 2; ++i)
#pragma unroll
    for (int j = 0; j < 4; ++j) acc[i][j] = zero;

  stage(0, 0);
  for (int k = 0; k < NSTEP; ++k) {
    const int cur = k & 1;
    wait_async0();        // my stage(cur) chunks have landed in LDS
    __syncthreads();      // publish cur; all waves done reading 1-cur
    if (k + 1 < NSTEP) stage(1 - cur, (k + 1) * 32);

    v16bf af[2], bfr[4];
#pragma unroll
    for (int tm = 0; tm < 2; ++tm) {
      int r = wm * 32 + tm * 16 + ra;
      af[tm] = ld_frag(&As[cur][r * 32 + kg], &As[cur][r * 32 + kg + 16]);
    }
#pragma unroll
    for (int tn = 0; tn < 4; ++tn) {
      int r = wn * 64 + tn * 16 + ra;
      bfr[tn] = ld_frag(&Bs[cur][r * 32 + kg], &Bs[cur][r * 32 + kg + 16]);
    }
#pragma unroll
    for (int tm = 0; tm < 2; ++tm)
#pragma unroll
      for (int tn = 0; tn < 4; ++tn)
        acc[tm][tn] = WMMA_BF16(af[tm], bfr[tn], acc[tm][tn]);
  }

  // epilogue: bias, (Q scale), scatter into attention-friendly layouts
#pragma unroll
  for (int tm = 0; tm < 2; ++tm)
#pragma unroll
    for (int tn = 0; tn < 4; ++tn) {
      int nb = n0 + wn * 64 + tn * 16 + ra;
      float bv_ = bias[nb];
      int h = nb >> 6, d = nb & 63;
#pragma unroll
      for (int r = 0; r < 8; ++r) {
        int m = m0 + wm * 32 + tm * 16 + r + 8 * half;
        int b = m >> 11, s = m & 2047;
        float v = acc[tm][tn][r] + bv_;
        if (z == 0)
          Qw[((size_t)(b * 16 + h) * 2048 + s) * 64 + d] = (bf16)(v * 0.125f);
        else if (z == 1)
          Kw[((size_t)(b * 16 + h) * 2048 + s) * 64 + d] = (bf16)v;
        else
          Vtw[((size_t)(b * 16 + h) * 64 + d) * 2048 + s] = (bf16)v;
      }
    }
}

// --------------------------------------------------- flash attention core
__global__ __launch_bounds__(128)
void attn_kernel(const bf16* __restrict__ Qw, const bf16* __restrict__ Kw,
                 const bf16* __restrict__ Vtw, float* __restrict__ Out) {
  constexpr int S = 2048, D = 64;
  constexpr int NKB = S / 64;
  const int bh = blockIdx.y;
  const int q0 = blockIdx.x * 64;
  const bf16* Qp = Qw  + (size_t)bh * S * D;
  const bf16* Kp = Kw  + (size_t)bh * S * D;
  const bf16* Vp = Vtw + (size_t)bh * D * S;  // V^T: [D][S]

  const int tid  = threadIdx.x;
  const int wave = tid >> 5;
  const int lane = tid & 31;
  const int ra   = lane & 15;
  const int kg   = (lane >> 4) * 8;
  const int half = lane >> 4;

  __shared__ __align__(16) bf16 Ks[2][64 * 64];   // 16 KB
  __shared__ __align__(16) bf16 Vs[2][64 * 64];   // 16 KB
  __shared__ __align__(16) bf16 Ps[4][16 * 64];   //  8 KB

  auto stageKV = [&](int buf, int s0) {
#pragma unroll
    for (int i = 0; i < 4; ++i) {  // 512 chunks each for K and V^T, 4+4/thread
      int c = tid + 128 * i;
      int row = c >> 3, ch = c & 7;
      stage16(&Ks[buf][row * 64 + ch * 8], &Kp[(size_t)(s0 + row) * D + ch * 8]);
      stage16(&Vs[buf][row * 64 + ch * 8], &Vp[(size_t)row * S + s0 + ch * 8]);
    }
  };

  // Q fragments resident in registers for the whole kernel (pre-scaled by 1/8)
  v16bf qf[2];
#pragma unroll
  for (int kk = 0; kk < 2; ++kk) {
    const bf16* p = &Qp[(size_t)(q0 + wave * 16 + ra) * D + kk * 32 + kg];
    qf[kk] = ld_frag(p, p + 16);
  }

  v8f zero = {};
  v8f o[4];
#pragma unroll
  for (int j = 0; j < 4; ++j) o[j] = zero;
  float mrun[8], lrun[8];
#pragma unroll
  for (int e = 0; e < 8; ++e) { mrun[e] = -1e30f; lrun[e] = 0.f; }

  stageKV(0, 0);
  for (int kb = 0; kb < NKB; ++kb) {
    const int cur = kb & 1;
    wait_async0();        // my stageKV(cur) chunks landed
    __syncthreads();      // publish cur; all waves done with 1-cur
    if (kb + 1 < NKB) stageKV(1 - cur, (kb + 1) * 64);

    // S = Q K^T (A=Q rows striped, B=K rows striped: identical load pattern)
    v8f sacc[4];
#pragma unroll
    for (int j = 0; j < 4; ++j) sacc[j] = zero;
#pragma unroll
    for (int kk = 0; kk < 2; ++kk)
#pragma unroll
      for (int j = 0; j < 4; ++j) {
        const bf16* p = &Ks[cur][(j * 16 + ra) * 64 + kk * 32 + kg];
        v16bf kf = ld_frag(p, p + 16);
        sacc[j] = WMMA_BF16(qf[kk], kf, sacc[j]);
      }

    // online softmax: C-layout row = e + 8*half, cols across 16-lane half
    float mnew[8], scl[8];
#pragma unroll
    for (int e = 0; e < 8; ++e) {
      float t = fmaxf(fmaxf(sacc[0][e], sacc[1][e]), fmaxf(sacc[2][e], sacc[3][e]));
      t = fmaxf(t, __shfl_xor(t, 1, 32));
      t = fmaxf(t, __shfl_xor(t, 2, 32));
      t = fmaxf(t, __shfl_xor(t, 4, 32));
      t = fmaxf(t, __shfl_xor(t, 8, 32));
      mnew[e] = fmaxf(mrun[e], t);
      scl[e]  = __expf(mrun[e] - mnew[e]);
    }
    float rsum[8];
#pragma unroll
    for (int e = 0; e < 8; ++e) rsum[e] = 0.f;
#pragma unroll
    for (int j = 0; j < 4; ++j)
#pragma unroll
      for (int e = 0; e < 8; ++e) {
        float p = __expf(sacc[j][e] - mnew[e]);
        sacc[j][e] = p;
        rsum[e] += p;
      }
#pragma unroll
    for (int e = 0; e < 8; ++e) {
      float rs = rsum[e];
      rs += __shfl_xor(rs, 1, 32);
      rs += __shfl_xor(rs, 2, 32);
      rs += __shfl_xor(rs, 4, 32);
      rs += __shfl_xor(rs, 8, 32);
      lrun[e] = lrun[e] * scl[e] + rs;
      mrun[e] = mnew[e];
    }
#pragma unroll
    for (int j = 0; j < 4; ++j)
#pragma unroll
      for (int e = 0; e < 8; ++e) o[j][e] *= scl[e];

    // re-stripe P: C-layout -> A-layout via LDS
#pragma unroll
    for (int j = 0; j < 4; ++j)
#pragma unroll
      for (int e = 0; e < 8; ++e)
        Ps[wave][(e + 8 * half) * 64 + j * 16 + ra] = (bf16)sacc[j][e];
    __syncthreads();

    // O += P @ V  (B-frag = contiguous V^T rows)
#pragma unroll
    for (int kk = 0; kk < 2; ++kk) {
      const bf16* pa = &Ps[wave][ra * 64 + kk * 32 + kg];
      v16bf pf = ld_frag(pa, pa + 16);
#pragma unroll
      for (int j = 0; j < 4; ++j) {
        const bf16* pb = &Vs[cur][(j * 16 + ra) * 64 + kk * 32 + kg];
        v16bf vf = ld_frag(pb, pb + 16);
        o[j] = WMMA_BF16(pf, vf, o[j]);
      }
    }
  }

  // epilogue: O / l, merge heads -> [B,S,E] fp32
  const int b = bh >> 4, h = bh & 15;
#pragma unroll
  for (int j = 0; j < 4; ++j) {
    int col = h * 64 + j * 16 + ra;
#pragma unroll
    for (int e = 0; e < 8; ++e) {
      int row = q0 + wave * 16 + e + 8 * half;
      Out[((size_t)(b * 2048 + row)) * 1024 + col] = o[j][e] / lrun[e];
    }
  }
}

// ----------------------------------------------------------------- launch
extern "C" void kernel_launch(void* const* d_in, const int* in_sizes, int n_in,
                              void* d_out, int out_size, void* d_ws, size_t ws_size,
                              hipStream_t stream) {
  const float* X  = (const float*)d_in[0];
  const float* Wq = (const float*)d_in[1];
  const float* bq = (const float*)d_in[2];
  const float* Wk = (const float*)d_in[3];
  const float* bk = (const float*)d_in[4];
  const float* Wv = (const float*)d_in[5];
  const float* bv = (const float*)d_in[6];
  float* out = (float*)d_out;

  char* ws = (char*)d_ws;
  bf16* Xbf = (bf16*)(ws);                           // 16 MB  [8192,1024]
  bf16* Wt  = (bf16*)(ws + ((size_t)16 << 20));      //  6 MB  3x W^T [1024,1024]
  bf16* Qw  = (bf16*)(ws + ((size_t)22 << 20));      // 16 MB  [B,H,S,D]
  bf16* Kw  = (bf16*)(ws + ((size_t)38 << 20));      // 16 MB  [B,H,S,D]
  bf16* Vtw = (bf16*)(ws + ((size_t)54 << 20));      // 16 MB  [B,H,D,S]

  cvt_x_kernel<<<8192, 256, 0, stream>>>(X, Xbf, 8192 * 1024);

  dim3 gw(32, 32, 3);
  cvt_w_kernel<<<gw, 256, 0, stream>>>(Wq, Wk, Wv, Wt);

  dim3 gg(4, 128, 3);  // N/256, M/64, {q,k,v}
  qkv_gemm_kernel<<<gg, 256, 0, stream>>>(Xbf, Wt, bq, bk, bv, Qw, Kw, Vtw);

  dim3 ga(32, 64, 1);  // S/64 q-blocks, B*H
  attn_kernel<<<ga, 128, 0, stream>>>(Qw, Kw, Vtw, out);
}
